// L1OLoss_51488067944955
// MI455X (gfx1250) — matrix operand order repaired and checked by reading, not compiled, not run on hardware
//
#include <hip/hip_runtime.h>
#include <hip/hip_bf16.h>
#include <math.h>

// ---------------- problem constants (from reference) ----------------
#define KS   17
#define NK   32
#define BAT  4
#define H    512
#define W    512
#define HP   544          // H + 16 (conv rows) + 16 (K-padding over-read rows), zero-filled
#define WP   576          // W + 2*PAD + slack for K over-read columns, zero-filled
#define NCHUNK 10         // K chunks: 8 x (2 rows x 16 cols) + row16 + col16
#define PI_F 3.14159265358979323846f

// ---------------- workspace layout (bytes) ----------------
#define ACC_OFF 0         // 4 floats: absSum, logSum, maskSum, pad
#define WB_OFF  64        // NCHUNK*32*32 bf16 weight bank = 20480 B
#define GP_OFF  20544     // BAT*HP*WP bf16 padded gray    = 2506752 B

typedef __attribute__((ext_vector_type(16))) __bf16 v16bf;
typedef __attribute__((ext_vector_type(8)))  float  v8f;
typedef __attribute__((ext_vector_type(4)))  unsigned int u32x4;
typedef __attribute__((ext_vector_type(8)))  int i32x8;
typedef __attribute__((ext_vector_type(4)))  int i32x4;

#if defined(__has_builtin)
#if __has_builtin(__builtin_amdgcn_tensor_load_to_lds) && \
    __has_builtin(__builtin_amdgcn_s_wait_tensorcnt)
#define HAVE_TDM 1
#endif
#endif

__device__ __forceinline__ unsigned short f2bf(float f) {
  union { float f; unsigned int u; } c; c.f = f;
  unsigned int r = (c.u + 0x7FFFu + ((c.u >> 16) & 1u)) >> 16;
  return (unsigned short)r;
}

__device__ __forceinline__ float gabor_w(int r, int c, int n) {
  // reference: x along kernel-H axis, y along kernel-W axis
  float x  = (float)(r - 8);
  float y  = (float)(c - 8);
  float th = PI_F * (float)n / (float)NK;
  float ct = __cosf(th), st = __sinf(th);
  float xt =  x * ct + y * st;
  float yt = -x * st + y * ct;
  return __expf(-0.5f * (xt * xt * 0.25f + yt * yt * (1.0f / 9.0f))) *
         __cosf((PI_F * 0.5f) * xt);            // 2*pi/Lambda, Lambda = 4
}

// ---- kernel 1: build repacked Gabor bank + zero accumulators ----
// layout wb[chunk][n][k], k in [0,32):
//   chunk<8 : tap (row = 2*chunk + (k>=16), col = k&15)
//   chunk==8: tap (row 16, col k) for k<17, else 0
//   chunk==9: tap (row k, col 16) for k<16, else 0
__global__ void gabor_init_kernel(unsigned short* __restrict__ wb,
                                  float* __restrict__ accf) {
  int t = blockIdx.x * blockDim.x + threadIdx.x;
  if (t < 4) accf[t] = 0.0f;
  if (t >= NCHUNK * NK * NK) return;
  int chunk = t >> 10;
  int n = (t >> 5) & 31;
  int k = t & 31;
  float v = 0.0f;
  if (chunk < 8) {
    v = gabor_w(2 * chunk + (k >> 4), k & 15, n);
  } else if (chunk == 8) {
    if (k < KS) v = gabor_w(16, k, n);
  } else {
    if (k < 16) v = gabor_w(k, 16, n);
  }
  wb[t] = f2bf(v);
}

// ---- kernel 2: fused *255 + RGB->gray + zero-pad, stored bf16 ----
__global__ void gray_pad_kernel(const float* __restrict__ im,
                                unsigned short* __restrict__ gp) {
  int t = blockIdx.x * blockDim.x + threadIdx.x;
  if (t >= BAT * HP * WP) return;
  int x = t % WP;
  int y = (t / WP) % HP;
  int b = t / (WP * HP);
  float g = 0.0f;
  int iy = y - 8, ix = x - 8;
  if (iy >= 0 && iy < H && ix >= 0 && ix < W) {
    size_t p = (((size_t)b * 3) * H + iy) * W + ix;
    float r  = im[p];
    float gg = im[p + (size_t)H * W];
    float bb = im[p + 2 * (size_t)H * W];
    g = 255.0f * (0.299f * r + 0.587f * gg + 0.144f * bb);  // reference uses 0.144
  }
  gp[t] = f2bf(g);
}

// ---- kernel 3: WMMA implicit-GEMM conv + fused argmax/orientation/loss epilogue ----
// block = 256 threads = 8 wave32; each wave: 16 output pixels (one row) x 32 kernels.
__global__ __launch_bounds__(256) void gabor_conv_kernel(
    const unsigned short* __restrict__ gp,   // padded gray bf16
    const unsigned short* __restrict__ wb,   // weight bank bf16 [10][32][32]
    const float* __restrict__ ref,           // (B,2,H,W)
    const float* __restrict__ mask,          // (B,1,H,W)
    float* __restrict__ out_orient,          // (B,2,H,W)  (= d_out + 2)
    float* __restrict__ accf) {
  __shared__ __align__(16) unsigned char smem[NCHUNK * NK * NK * 2];  // 20480 B
  __shared__ float sred[3];

  const int tid  = threadIdx.x;
  const int bid  = blockIdx.x;                 // BAT * H * (W/128) blocks
  const int xblk = bid & 3;
  const int yy   = (bid >> 2) & (H - 1);
  const int b    = bid >> 11;
  const int x0b  = xblk * 128;
  const int wave = tid >> 5;
  const int lane = tid & 31;
  const int n    = lane & 15;
  const int hi   = lane >> 4;
  const int kb   = hi << 3;                    // per-lane K sub-offset (0 or 8)

  // ---- stage weight bank into LDS via Tensor Data Mover (wave 0 only; EXEC ignored)
#if HAVE_TDM
  if (wave == 0) {
    unsigned long long ga = (unsigned long long)(uintptr_t)wb;
    unsigned int lds = (unsigned int)(uintptr_t)&smem[0];   // low 32 bits = LDS offset
    u32x4 g0;
    g0.x = 0x1u;                                            // count=1, user mode
    g0.y = lds;                                             // lds_addr
    g0.z = (unsigned int)ga;                                // global_addr[31:0]
    g0.w = (unsigned int)((ga >> 32) & 0x01FFFFFFu) | 0x80000000u;  // addr[56:32] | type=2
    i32x8 g1;
    g1[0] = 0x00030000;               // workgroup_mask=0, data_size=3 (8 bytes)
    g1[1] = (int)(2560u << 16);       // tensor_dim0 = 2560 (20480 B / 8)
    g1[2] = (int)(1u << 16);          // tensor_dim1 = 1
    g1[3] = (int)(2560u << 16);       // tile_dim0  = 2560
    g1[4] = 1;                        // tile_dim1  = 1
    g1[5] = 2560;                     // tensor_dim0_stride = 2560
    g1[6] = 0;
    g1[7] = 0;
    i32x4 z4 = {0, 0, 0, 0};
#if __clang_major__ >= 23
    i32x8 z8 = {0, 0, 0, 0, 0, 0, 0, 0};
    __builtin_amdgcn_tensor_load_to_lds(g0, g1, z4, z4, z8, 0);
#else
    __builtin_amdgcn_tensor_load_to_lds(g0, g1, z4, z4, 0);
#endif
    __builtin_amdgcn_s_wait_tensorcnt(0);
  }
#else
  {
    const unsigned int* src = (const unsigned int*)wb;
    unsigned int* dst = (unsigned int*)smem;
    #pragma unroll 4
    for (int i = tid; i < (NCHUNK * NK * NK) / 2; i += 256) dst[i] = src[i];
  }
#endif
  __syncthreads();

  // A-fragment base: lane row M = n; window starts at column x0w + M
  const int x0w = x0b + wave * 16;
  const unsigned short* grow0 =
      gp + ((size_t)(b * HP + yy)) * WP + (x0w + n);
  const unsigned short* wsm = (const unsigned short*)smem;

  v8f acc0 = {};   // N = 0..15
  v8f acc1 = {};   // N = 16..31

  // ---- chunks 0..7: K=32 = (row 2t, cols 0..15) ++ (row 2t+1, cols 0..15)
  #pragma unroll
  for (int t = 0; t < 8; ++t) {
    const unsigned short* r0p = grow0 + (size_t)(2 * t) * WP;
    const unsigned short* r1p = r0p + WP;
    union { unsigned short u[16]; v16bf v; } a;
    __builtin_memcpy(&a.u[0], r0p + kb, 16);   // K 0..15 half -> row 2t
    __builtin_memcpy(&a.u[8], r1p + kb, 16);   // K 16..31 half -> row 2t+1
    union { unsigned short u[16]; v16bf v; } b0, b1;
    __builtin_memcpy(&b0.u[0], wsm + (((t << 5) + n) << 5) + kb, 16);
    __builtin_memcpy(&b0.u[8], wsm + (((t << 5) + n) << 5) + 16 + kb, 16);
    __builtin_memcpy(&b1.u[0], wsm + (((t << 5) + 16 + n) << 5) + kb, 16);
    __builtin_memcpy(&b1.u[8], wsm + (((t << 5) + 16 + n) << 5) + 16 + kb, 16);
    acc0 = __builtin_amdgcn_wmma_f32_16x16x32_bf16(false, a.v, false, b0.v,
                                                   (short)0, acc0, false, false);
    acc1 = __builtin_amdgcn_wmma_f32_16x16x32_bf16(false, a.v, false, b1.v,
                                                   (short)0, acc1, false, false);
    if (t < 7) __builtin_prefetch(r0p + 2 * WP, 0, 1);   // next row pair
  }
  // ---- chunk 8: row 16, cols 0..16 (K padded to 32; pad cols are zero-filled)
  {
    const unsigned short* rp = grow0 + (size_t)16 * WP;
    union { unsigned short u[16]; v16bf v; } a;
    __builtin_memcpy(&a.u[0], rp + kb, 16);
    __builtin_memcpy(&a.u[8], rp + 16 + kb, 16);
    union { unsigned short u[16]; v16bf v; } b0, b1;
    __builtin_memcpy(&b0.u[0], wsm + (((8 << 5) + n) << 5) + kb, 16);
    __builtin_memcpy(&b0.u[8], wsm + (((8 << 5) + n) << 5) + 16 + kb, 16);
    __builtin_memcpy(&b1.u[0], wsm + (((8 << 5) + 16 + n) << 5) + kb, 16);
    __builtin_memcpy(&b1.u[8], wsm + (((8 << 5) + 16 + n) << 5) + 16 + kb, 16);
    acc0 = __builtin_amdgcn_wmma_f32_16x16x32_bf16(false, a.v, false, b0.v,
                                                   (short)0, acc0, false, false);
    acc1 = __builtin_amdgcn_wmma_f32_16x16x32_bf16(false, a.v, false, b1.v,
                                                   (short)0, acc1, false, false);
  }
  // ---- chunk 9: col 16, rows 0..15 (strided; over-read rows are zero-filled)
  {
    union { unsigned short u[16]; v16bf v; } a;
    #pragma unroll
    for (int i = 0; i < 8; ++i) {
      a.u[i]     = grow0[(size_t)(kb + i) * WP + 16];
      a.u[8 + i] = grow0[(size_t)(16 + kb + i) * WP + 16];  // x0 weights (rows>=16 K-pad)
    }
    union { unsigned short u[16]; v16bf v; } b0, b1;
    __builtin_memcpy(&b0.u[0], wsm + (((9 << 5) + n) << 5) + kb, 16);
    __builtin_memcpy(&b0.u[8], wsm + (((9 << 5) + n) << 5) + 16 + kb, 16);
    __builtin_memcpy(&b1.u[0], wsm + (((9 << 5) + 16 + n) << 5) + kb, 16);
    __builtin_memcpy(&b1.u[8], wsm + (((9 << 5) + 16 + n) << 5) + 16 + kb, 16);
    acc0 = __builtin_amdgcn_wmma_f32_16x16x32_bf16(false, a.v, false, b0.v,
                                                   (short)0, acc0, false, false);
    acc1 = __builtin_amdgcn_wmma_f32_16x16x32_bf16(false, a.v, false, b1.v,
                                                   (short)0, acc1, false, false);
  }
  __syncthreads();   // all waves done with LDS weights; reuse LDS for scores

  // scatter accumulators: scores[pixel(0..127)][kernel(0..31)]
  float* sc = (float*)smem;
  #pragma unroll
  for (int j = 0; j < 8; ++j) {
    int M = j + (hi << 3);                    // C/D layout: vgpr j, lane-half hi
    sc[(((wave << 4) + M) << 5) + n]      = acc0[j];
    sc[(((wave << 4) + M) << 5) + 16 + n] = acc1[j];
  }
  if (tid < 3) sred[tid] = 0.0f;
  __syncthreads();

  // per-pixel epilogue: ReLU + argmax + tanh-conf + sin/cos + loss partials
  if (tid < 128) {
    int x = x0b + tid;
    const float4* row = (const float4*)(sc + (tid << 5));
    float best = -1.0f; int bidx = 0;
    #pragma unroll
    for (int q = 0; q < 8; ++q) {
      float4 v4 = row[q];
      float vv[4] = {v4.x, v4.y, v4.z, v4.w};
      #pragma unroll
      for (int c = 0; c < 4; ++c) {
        float v = fmaxf(vv[c], 0.0f);
        int idx = q * 4 + c;
        if (v > best) { best = v; bidx = idx; }
      }
    }
    // tanh(best) for best >= 0 via fast exp
    float e2 = __expf(-2.0f * best);
    float conf = 0.5f * ((1.0f - e2) / (1.0f + e2) + 1.0f);
    float th2  = (float)bidx * (2.0f * PI_F / (float)NK);
    float s = __sinf(th2) * conf;
    float c = __cosf(th2) * conf;

    size_t o0 = (((size_t)b * 2) * H + yy) * W + x;
    size_t o1 = o0 + (size_t)H * W;
    out_orient[o0] = s;
    out_orient[o1] = c;

    float mk = mask[((size_t)b * H + yy) * W + x];
    float r0 = ref[o0], r1 = ref[o1];
    float pa = fabsf(s * mk - r0 * mk) + fabsf(c * mk - r1 * mk);
    float cc = fminf(fmaxf(conf, 0.001f), 1.0f);
    float pc = -__logf(cc) * mk;

    atomicAdd(&sred[0], pa);   // ds_add_f32
    atomicAdd(&sred[1], pc);
    atomicAdd(&sred[2], mk);
  }
  __syncthreads();
  if (tid == 0) {
    atomicAdd(&accf[0], sred[0]);   // global_atomic_add_f32
    atomicAdd(&accf[1], sred[1]);
    atomicAdd(&accf[2], sred[2]);
  }
}

// ---- kernel 4: finalize scalar losses ----
__global__ void finalize_kernel(const float* __restrict__ accf,
                                float* __restrict__ out) {
  if (blockIdx.x == 0 && threadIdx.x == 0) {
    out[0] = accf[0] / (float)((size_t)BAT * 2 * H * W);  // mean |.|
    out[1] = accf[1] / accf[2];                           // -sum(log conf * m)/sum(m)
  }
}

extern "C" void kernel_launch(void* const* d_in, const int* in_sizes, int n_in,
                              void* d_out, int out_size, void* d_ws, size_t ws_size,
                              hipStream_t stream) {
  (void)in_sizes; (void)n_in; (void)out_size; (void)ws_size;
  const float* gen  = (const float*)d_in[0];   // (4,3,512,512) f32
  const float* ref  = (const float*)d_in[1];   // (4,2,512,512) f32
  const float* mask = (const float*)d_in[2];   // (4,1,512,512) f32
  float* out = (float*)d_out;                  // [loss0, loss1, orientation_fake...]

  unsigned char* ws = (unsigned char*)d_ws;
  float*          accf = (float*)(ws + ACC_OFF);
  unsigned short* wb   = (unsigned short*)(ws + WB_OFF);
  unsigned short* gp   = (unsigned short*)(ws + GP_OFF);

  gabor_init_kernel<<<(NCHUNK * NK * NK + 255) / 256, 256, 0, stream>>>(wb, accf);
  gray_pad_kernel<<<(BAT * HP * WP + 255) / 256, 256, 0, stream>>>(gen, gp);
  gabor_conv_kernel<<<BAT * H * (W / 128), 256, 0, stream>>>(gp, wb, ref, mask,
                                                             out + 2, accf);
  finalize_kernel<<<1, 64, 0, stream>>>(accf, out);
}
